// Glm4MoeTopkRouter_1657857376738
// MI455X (gfx1250) — matrix operand order repaired and checked by reading, not compiled, not run on hardware
//
#include <hip/hip_runtime.h>
#include <hip/hip_bf16.h>

// ---------------------------------------------------------------------------
// GLM4-MoE top-k router for MI455X (gfx1250, wave32, WMMA).
//
// router_logits = x[T,4096] @ W^T[4096,128]  (34.4 GFLOP; x = 512MB read-once
// -> 22us HBM floor @23.3TB/s; W = 2MB, L2-resident). fp32 VALU would be the
// bottleneck, so use v_wmma_f32_16x16x32_bf16 with hi/lo bf16 split
// (hi*hi + hi*lo + lo*hi ~ fp32 precision) on THREE independent accumulators
// to avoid back-to-back WMMA RAW stalls.
// ---------------------------------------------------------------------------

typedef __attribute__((ext_vector_type(16))) __bf16 v16bf;
typedef __attribute__((ext_vector_type(8)))  float  v8f;

#define HIDDEN   4096
#define NEXP     128
#define TOPK     8
#define NGROUP   8
#define TOPKG    4
#define KC       128    // K-chunk staged in LDS per iteration (4 WMMA substeps)

union ABFrag { v16bf v; uint4 q[2]; };

// ---------------------------------------------------------------------------
// Kernel 0: split fp32 weights into bf16 hi/lo, pre-swizzled into the exact
// contiguous-k per-lane B-fragment layout shared with the A reads (B 32x16: lanes 0-15 hold K=0..15,
// lanes 16-31 hold K=16..31, K-pairs packed per dword).
// ---------------------------------------------------------------------------
__global__ __launch_bounds__(256) void router_prep_weights(
    const float* __restrict__ weight,
    __bf16* __restrict__ wsH, __bf16* __restrict__ wsL)
{
    int id   = blockIdx.x * 256 + threadIdx.x;   // 0 .. 524287
    int j    = id & 15;
    int lane = (id >> 4) & 31;
    int s    = (id >> 9) & 127;                  // k-step (32 wide)
    int et   = id >> 16;                         // expert tile (0..7)

    int e = et * 16 + (lane & 15);
    int k = s * 32 + ((lane >> 4) << 4) + j;

    float  w  = weight[(size_t)e * HIDDEN + k];
    __bf16 hi = (__bf16)w;
    __bf16 lo = (__bf16)(w - (float)hi);

    size_t fi = ((size_t)(et * 128 + s) * 32 + lane) * 16 + j;
    wsH[fi] = hi;
    wsL[fi] = lo;
}

// ---------------------------------------------------------------------------
// Kernel 1: fused GEMM (WMMA bf16x2) + sigmoid + wave-parallel grouped top-k.
// Block = 16 tokens, 256 threads = 8 waves; wave w owns expert tile w.
// ---------------------------------------------------------------------------
__global__ __launch_bounds__(256) void router_main(
    const float* __restrict__ x,
    const float* __restrict__ bias,
    const __bf16* __restrict__ wsH,
    const __bf16* __restrict__ wsL,
    int*   __restrict__ out_idx,
    float* __restrict__ out_w)
{
    __shared__ __bf16 ah[16][KC + 8];   // stride 136 -> conflict-free b128 reads
    __shared__ __bf16 al[16][KC + 8];
    __shared__ float  slog[16][129];    // logits -> raw sigmoid scores
    __shared__ float  scor[16][129];    // bias-corrected scores
    __shared__ float  sbias[NEXP];

    const int tid  = threadIdx.x;
    const int lane = tid & 31;
    const int et   = tid >> 5;                  // wave id = expert tile
    const int tok0 = blockIdx.x * 16;

    if (tid < NEXP) sbias[tid] = bias[tid];

    // cooperative A loader: 16 threads/row, two float4 per thread per chunk
    const int arow = tid >> 4;
    const int acol = (tid & 15) << 2;
    const float* aptr = x + (size_t)(tok0 + arow) * HIDDEN + acol;

    // WMMA fragment coordinates
    const int m    = lane & 15;
    const int off0 = (lane < 16) ? 0 : 16;
    const uint4* bhq = (const uint4*)wsH + ((size_t)et * 128 * 32 + lane) * 2;
    const uint4* blq = (const uint4*)wsL + ((size_t)et * 128 * 32 + lane) * 2;

    v8f cH = {}, cM = {}, cL = {};   // independent accumulators: hh, hl, lh

    for (int k0 = 0; k0 < HIDDEN; k0 += KC) {
        // ---- stage x chunk into LDS as bf16 hi/lo split ----
        float4 xa = *(const float4*)(aptr + k0);
        float4 xb = *(const float4*)(aptr + k0 + 64);
        if (k0 + KC < HIDDEN) {
            __builtin_prefetch(aptr + k0 + KC, 0, 1);       // global_prefetch_b8
            __builtin_prefetch(aptr + k0 + KC + 64, 0, 1);
        }
        {
            float vx[8] = {xa.x, xa.y, xa.z, xa.w, xb.x, xb.y, xb.z, xb.w};
            #pragma unroll
            for (int j = 0; j < 8; ++j) {
                int cc = (j < 4) ? (acol + j) : (acol + 60 + j);  // +64 half
                __bf16 h = (__bf16)vx[j];
                ah[arow][cc] = h;
                al[arow][cc] = (__bf16)(vx[j] - (float)h);
            }
        }
        __syncthreads();

        // ---- 4 substeps of K=32; 3 WMMAs each on separate accumulators ----
        #pragma unroll
        for (int s = 0; s < KC / 32; ++s) {
            const int kk = s * 32;
            ABFrag fah, fal, fbh, fbl;
            fah.q[0] = *(const uint4*)&ah[m][kk + off0];
            fah.q[1] = *(const uint4*)&ah[m][kk + off0 + 8];
            fal.q[0] = *(const uint4*)&al[m][kk + off0];
            fal.q[1] = *(const uint4*)&al[m][kk + off0 + 8];

            const size_t sidx = (size_t)(k0 >> 5) + s;
            const uint4* bh = bhq + sidx * 64;
            const uint4* bl = blq + sidx * 64;
            fbh.q[0] = bh[0]; fbh.q[1] = bh[1];
            fbl.q[0] = bl[0]; fbl.q[1] = bl[1];

            cH = __builtin_amdgcn_wmma_f32_16x16x32_bf16(
                     false, fah.v, false, fbh.v, (short)0, cH, false, false);
            cM = __builtin_amdgcn_wmma_f32_16x16x32_bf16(
                     false, fah.v, false, fbl.v, (short)0, cM, false, false);
            cL = __builtin_amdgcn_wmma_f32_16x16x32_bf16(
                     false, fal.v, false, fbh.v, (short)0, cL, false, false);
        }
        __syncthreads();
    }

    // ---- combine accumulators, scatter C tile to LDS ----
    {
        v8f c = cH + cM + cL;
        const int mb = (lane < 16) ? 0 : 8;
        const int n  = et * 16 + (lane & 15);
        #pragma unroll
        for (int r = 0; r < 8; ++r)
            slog[mb + r][n] = c[r];
    }
    __syncthreads();

    // ---- parallel sigmoid + bias correction (2048 values / 256 threads) ----
    for (int i = tid; i < 16 * NEXP; i += 256) {
        int mm = i >> 7, e = i & 127;
        float v  = slog[mm][e];
        float sg = 1.0f / (1.0f + __expf(-v));
        slog[mm][e] = sg;                 // raw score (for weight gather)
        scor[mm][e] = sg + sbias[e];      // score-for-choice
    }
    __syncthreads();

    // =======================================================================
    // Wave-parallel grouped top-k: wave w handles tokens 2w, 2w+1.
    // 16 lanes per token; lane hl owns experts [hl*8, hl*8+8) -> all in
    // group (hl>>1). Everything in registers + shfl; no more LDS scans.
    // =======================================================================
    {
        const int half = lane >> 4;            // 0/1 -> which token
        const int hl   = lane & 15;            // lane-in-token
        const int mm   = 2 * et + half;        // token row in block
        const int hbase = half << 4;           // first abs lane of this half

        float cv[8], rv[8];
        #pragma unroll
        for (int j = 0; j < 8; ++j) {
            cv[j] = scor[mm][hl * 8 + j];
            rv[j] = slog[mm][hl * 8 + j];
        }

        // group score = sum of top-2 corrected within group of 16 (2 lanes)
        float m1 = -1e30f, m2 = -1e30f;
        #pragma unroll
        for (int j = 0; j < 8; ++j) {
            float v = cv[j];
            if (v > m1)      { m2 = m1; m1 = v; }
            else if (v > m2) { m2 = v; }
        }
        {
            float o1 = __shfl_xor(m1, 1);
            float o2 = __shfl_xor(m2, 1);
            float n1 = fmaxf(m1, o1);
            float n2 = fmaxf(fminf(m1, o1), fmaxf(m2, o2));
            m1 = n1; m2 = n2;                  // both lanes of pair hold top-2
        }
        const float gs = m1 + m2;

        // broadcast all 8 group scores to every lane (unrolled shuffles)
        float garr[8];
        #pragma unroll
        for (int g = 0; g < 8; ++g)
            garr[g] = __shfl(gs, hbase + 2 * g);

        // top-4 groups (every lane computes identical mask)
        unsigned gmask = 0;
        #pragma unroll
        for (int it = 0; it < TOPKG; ++it) {
            float best = -1e30f; int bg = 0;
            #pragma unroll
            for (int g = 0; g < NGROUP; ++g) {
                if ((gmask >> g) & 1u) continue;
                if (garr[g] > best) { best = garr[g]; bg = g; }
            }
            gmask |= 1u << bg;
        }

        // masked values (reference: inactive groups contribute 0.0)
        const bool act = (gmask >> (hl >> 1)) & 1u;
        float mv[8];
        #pragma unroll
        for (int j = 0; j < 8; ++j) mv[j] = act ? cv[j] : 0.0f;

        // 8 selection rounds: lane-local argmax + 4-step shfl_xor reduce
        // (strict >, lower index wins ties -> matches jax.lax.top_k order)
        unsigned sel = 0;
        int   ridx[TOPK];
        float rraw[TOPK];
        float wsum = 0.0f;
        #pragma unroll
        for (int it = 0; it < TOPK; ++it) {
            float bv = -1e30f; int bi = 127; float br = 0.0f;
            #pragma unroll
            for (int j = 0; j < 8; ++j) {
                float v = ((sel >> j) & 1u) ? -1e30f : mv[j];
                if (v > bv) { bv = v; bi = hl * 8 + j; br = rv[j]; }
            }
            #pragma unroll
            for (int off = 1; off < 16; off <<= 1) {
                float ov = __shfl_xor(bv, off);
                int   oi = __shfl_xor(bi, off);
                float orr = __shfl_xor(br, off);
                if (ov > bv || (ov == bv && oi < bi)) { bv = ov; bi = oi; br = orr; }
            }
            ridx[it] = bi;
            rraw[it] = br;
            wsum += br;
            if ((bi >> 3) == hl) sel |= 1u << (bi & 7);   // owning lane marks it
        }

        if (hl == 0) {
            const float inv = 1.0f / (wsum + 1e-20f);     // NORM_TOPK, scale=1
            const size_t t = (size_t)(tok0 + mm);
            #pragma unroll
            for (int it = 0; it < TOPK; ++it) {
                out_idx[t * TOPK + it] = ridx[it];
                out_w  [t * TOPK + it] = rraw[it] * inv;
            }
        }
    }
}

// ---------------------------------------------------------------------------
extern "C" void kernel_launch(void* const* d_in, const int* in_sizes, int n_in,
                              void* d_out, int out_size, void* d_ws, size_t ws_size,
                              hipStream_t stream) {
    const float* x      = (const float*)d_in[0];   // [T, 4096]
    const float* weight = (const float*)d_in[1];   // [128, 4096]
    const float* bias   = (const float*)d_in[2];   // [128]

    const int T = in_sizes[0] / HIDDEN;

    // workspace: pre-swizzled bf16 hi/lo weight fragments (1 MB + 1 MB)
    __bf16* wsH = (__bf16*)d_ws;
    __bf16* wsL = wsH + (size_t)NEXP * HIDDEN;

    // outputs: indices (int32) first, then weights (float), per ref order
    int*   out_idx = (int*)d_out;
    float* out_w   = (float*)d_out + (size_t)T * TOPK;

    router_prep_weights<<<(NEXP * HIDDEN) / 256, 256, 0, stream>>>(weight, wsH, wsL);
    router_main<<<T / 16, 256, 0, stream>>>(x, bias, wsH, wsL, out_idx, out_w);
}
